// HAN_23003844837681
// MI455X (gfx1250) — matrix operand order repaired and checked by reading, not compiled, not run on hardware
//
#include <hip/hip_runtime.h>

// ---------------------------------------------------------------------------
// Types for CDNA5 WMMA (gfx1250, wave32)
// ---------------------------------------------------------------------------
typedef __attribute__((ext_vector_type(16))) __bf16 v16bf;
typedef __attribute__((ext_vector_type(8)))  __bf16 v8bf;
typedef __attribute__((ext_vector_type(8)))  float  v8f;

static __device__ __forceinline__ __bf16 f2bf(float f) {
    unsigned u = __float_as_uint(f);
    u = (u + 0x7FFFu + ((u >> 16) & 1u)) >> 16;
    union { unsigned short s; __bf16 b; } y;
    y.s = (unsigned short)u;
    return y.b;
}
static __device__ __forceinline__ float bf2f(__bf16 b) {
    union { unsigned short s; __bf16 b; } y;
    y.b = b;
    return __uint_as_float(((unsigned)y.s) << 16);
}
static __device__ __forceinline__ float sigmoidf_(float x) {
    return 1.0f / (1.0f + __expf(-x));
}

static __device__ __forceinline__ v8f wmma_bf16(v16bf a, v16bf b, v8f c) {
    // D(f32 16x16) = A(16x32 bf16) * B(32x16 bf16) + C
    return __builtin_amdgcn_wmma_f32_16x16x32_bf16(
        false, a, false, b, (short)0, c, false, false);
}

// A-operand loader (16x32 bf16 tile). Layout (matches loadB): lanes 0-15 row M=lane hold
// K 0..15; lanes 16-31 same rows hold K 16..31.
// => per lane: two contiguous 8-elem (16B) chunks at e0+hl*16 and e0+hl*16+8.
static __device__ __forceinline__ v16bf loadA(const __bf16* base, int stride,
                                              int row, int e0, int hl) {
    const __bf16* p = base + (size_t)row * stride + e0 + hl * 16;
    v8bf lo = *(const v8bf*)p;
    v8bf hi = *(const v8bf*)(p + 8);
    v16bf r;
#pragma unroll
    for (int i = 0; i < 8; ++i) { r[i] = lo[i]; r[8 + i] = hi[i]; }
    return r;
}

// B-operand loader (32x16 bf16 tile). Layout (same as loadA, contiguous K): lanes 0-15 col N=lane hold
// K 0..15; lanes 16-31 hold K 16..31 => 16 contiguous elems (32B) per lane.
// wrow points at W[n][0] for this lane's column n (row-major, K contiguous).
static __device__ __forceinline__ v16bf loadB(const __bf16* wrow, int e0, int hl) {
    return *(const v16bf*)(wrow + e0 + hl * 16);
}

// ---------------------------------------------------------------------------
// Constants
// ---------------------------------------------------------------------------
#define SN 16   // sentences
#define BN 32   // batch
#define TN 64   // tokens
#define EN 256  // embed
#define HN 256  // hidden
#define NC 50

// ---------------------------------------------------------------------------
// fp32 -> bf16 convert / transpose-convert
// ---------------------------------------------------------------------------
__global__ void cvt_bf16(const float* __restrict__ s, __bf16* __restrict__ d, int n) {
    int i = blockIdx.x * blockDim.x + threadIdx.x;
    if (i < n) d[i] = f2bf(s[i]);
}
// dst[n][h] = src[h][n], 512x512
__global__ void transcvt_bf16(const float* __restrict__ s, __bf16* __restrict__ d) {
    int i = blockIdx.x * blockDim.x + threadIdx.x;
    if (i >= 512 * 512) return;
    int n = i >> 9, h = i & 511;
    d[i] = f2bf(s[h * 512 + n]);
}

// ---------------------------------------------------------------------------
// Embedding gather: x[S,B,T] -> xall[S,T,B,E] (bf16)
// ---------------------------------------------------------------------------
__global__ void gather_embed(const int* __restrict__ x,
                             const float* __restrict__ embed,
                             __bf16* __restrict__ xall) {
    int idx = blockIdx.x * blockDim.x + threadIdx.x; // S*T*B*(E/4) threads
    if (idx >= SN * TN * BN * (EN / 4)) return;
    int q = idx & 63;
    int b = (idx >> 6) & 31;
    int t = (idx >> 11) & 63;
    int s = idx >> 17;
    int token = x[((s * BN) + b) * TN + t];
    float4 v = *(const float4*)(embed + (size_t)token * EN + q * 4);
    __bf16* d = xall + (((size_t)(s * TN + t) * BN + b) * EN + q * 4);
    d[0] = f2bf(v.x); d[1] = f2bf(v.y); d[2] = f2bf(v.z); d[3] = f2bf(v.w);
}

// ---------------------------------------------------------------------------
// Bidirectional GRU scan kernel (one block = one (sequence, direction)).
// 512 threads = 16 waves; wave w owns hidden cols [16w,16w+16).
// xin: [nseq][T][B][IN] bf16, out: [nseq][T][B][512] bf16 (dir picks col half).
// Weights: Wih [768][IN] bf16, Whh [768][256] bf16 (row-major, K contiguous).
// ---------------------------------------------------------------------------
__global__ __launch_bounds__(512) void gru_kernel(
    const __bf16* __restrict__ xin_base,
    const __bf16* __restrict__ WihF, const __bf16* __restrict__ WihB,
    const __bf16* __restrict__ WhhF, const __bf16* __restrict__ WhhB,
    const float* __restrict__ bihF, const float* __restrict__ bhhF,
    const float* __restrict__ bihB, const float* __restrict__ bhhB,
    const float* __restrict__ h0,          // [2][B][256]
    __bf16* __restrict__ out_base,         // [nseq][T][B][512]
    int T, int in_ksteps) {
    const int s   = blockIdx.x >> 1;
    const int dir = blockIdx.x & 1;
    const int IN  = in_ksteps * 32;

    const __bf16* xin = xin_base + (size_t)s * T * BN * IN;
    const __bf16* Wih = dir ? WihB : WihF;
    const __bf16* Whh = dir ? WhhB : WhhF;
    const float*  bih = dir ? bihB : bihF;
    const float*  bhh = dir ? bhhB : bhhF;
    __bf16* outp = out_base + (size_t)s * T * BN * 512 + dir * HN;
    const float* h0p = h0 + (size_t)dir * BN * HN;

    __shared__ float  hF[BN][HN];      // fp32 master hidden state (32 KB)
    __shared__ __bf16 hB[BN * HN];     // bf16 copy for WMMA A     (16 KB)

    const int tid = threadIdx.x;
    for (int i = tid; i < BN * HN; i += 512) {
        float v = h0p[i];
        hF[i >> 8][i & 255] = v;
        hB[i] = f2bf(v);
    }
    __syncthreads();

    const int wave = tid >> 5;
    const int lane = tid & 31;
    const int hl   = lane >> 4;        // lane half (K split)
    const int ln   = lane & 15;
    const int hcol = wave * 16 + ln;   // this lane's hidden column (B op) /
                                       // output column (C op), identical map
    const float b_r  = bih[hcol] + bhh[hcol];
    const float b_z  = bih[HN + hcol] + bhh[HN + hcol];
    const float b_xn = bih[2 * HN + hcol];
    const float b_hn = bhh[2 * HN + hcol];

    // Per-lane weight row pointers (B operand: column n = gate_base + ln)
    const __bf16* wih_r = Wih + (size_t)(0 * HN + wave * 16 + ln) * IN;
    const __bf16* wih_z = Wih + (size_t)(1 * HN + wave * 16 + ln) * IN;
    const __bf16* wih_n = Wih + (size_t)(2 * HN + wave * 16 + ln) * IN;
    const __bf16* whh_r = Whh + (size_t)(0 * HN + wave * 16 + ln) * HN;
    const __bf16* whh_z = Whh + (size_t)(1 * HN + wave * 16 + ln) * HN;
    const __bf16* whh_n = Whh + (size_t)(2 * HN + wave * 16 + ln) * HN;

    for (int step = 0; step < T; ++step) {
        const int t = dir ? (T - 1 - step) : step;
        const __bf16* xrow = xin + (size_t)t * BN * IN;

        v8f a_r[2] = {}, a_z[2] = {}, a_xn[2] = {}, a_hn[2] = {};

        // ---- input GEMM: x_t[32,IN] x Wih^T[IN,768] -------------------
        for (int kk = 0; kk < in_ksteps; ++kk) {
            const int e0 = kk * 32;
            v16bf A0 = loadA(xrow, IN, 0 + ln, e0, hl);
            v16bf A1 = loadA(xrow, IN, 16 + ln, e0, hl);
            v16bf Br = loadB(wih_r, e0, hl);
            v16bf Bz = loadB(wih_z, e0, hl);
            v16bf Bn = loadB(wih_n, e0, hl);
            a_r[0]  = wmma_bf16(A0, Br, a_r[0]);
            a_r[1]  = wmma_bf16(A1, Br, a_r[1]);
            a_z[0]  = wmma_bf16(A0, Bz, a_z[0]);
            a_z[1]  = wmma_bf16(A1, Bz, a_z[1]);
            a_xn[0] = wmma_bf16(A0, Bn, a_xn[0]);
            a_xn[1] = wmma_bf16(A1, Bn, a_xn[1]);
        }
        // ---- recurrent GEMM: h[32,256] x Whh^T[256,768] ---------------
        for (int kk = 0; kk < 8; ++kk) {
            const int e0 = kk * 32;
            v16bf A0 = loadA(&hB[0], HN, 0 + ln, e0, hl);
            v16bf A1 = loadA(&hB[0], HN, 16 + ln, e0, hl);
            v16bf Br = loadB(whh_r, e0, hl);
            v16bf Bz = loadB(whh_z, e0, hl);
            v16bf Bn = loadB(whh_n, e0, hl);
            a_r[0]  = wmma_bf16(A0, Br, a_r[0]);
            a_r[1]  = wmma_bf16(A1, Br, a_r[1]);
            a_z[0]  = wmma_bf16(A0, Bz, a_z[0]);
            a_z[1]  = wmma_bf16(A1, Bz, a_z[1]);
            a_hn[0] = wmma_bf16(A0, Bn, a_hn[0]);
            a_hn[1] = wmma_bf16(A1, Bn, a_hn[1]);
        }

        // ---- GRU nonlinearity entirely in registers -------------------
        float hnew[2][8];
#pragma unroll
        for (int rt = 0; rt < 2; ++rt) {
#pragma unroll
            for (int i = 0; i < 8; ++i) {
                int row = rt * 16 + hl * 8 + i;   // C layout: VGPR i -> M=i / M=8+i
                float r = sigmoidf_(a_r[rt][i] + b_r);
                float z = sigmoidf_(a_z[rt][i] + b_z);
                float n = tanhf(a_xn[rt][i] + b_xn + r * (a_hn[rt][i] + b_hn));
                hnew[rt][i] = (1.0f - z) * n + z * hF[row][hcol];
            }
        }
        __syncthreads();  // all WMMA reads of hB done before rewrite
#pragma unroll
        for (int rt = 0; rt < 2; ++rt) {
#pragma unroll
            for (int i = 0; i < 8; ++i) {
                int row = rt * 16 + hl * 8 + i;
                float v = hnew[rt][i];
                hF[row][hcol] = v;
                __bf16 bv = f2bf(v);
                hB[row * HN + hcol] = bv;
                outp[(size_t)t * BN * 512 + (size_t)row * 512 + hcol] = bv;
            }
        }
        __syncthreads();  // h writes visible before next step's reads
    }
}

// ---------------------------------------------------------------------------
// Attention scoring: scores[r] = dot(tanh(rows[r] @ W + b), proj)
// rows: [nrows][512] bf16, Wt: [512][512] bf16 PRE-TRANSPOSED (Wt[n][h]).
// One block = 32 rows, 16 waves (wave w -> cols [32w,32w+32)).
// ---------------------------------------------------------------------------
__global__ __launch_bounds__(512) void attn_scores(
    const __bf16* __restrict__ rows, const __bf16* __restrict__ Wt,
    const float* __restrict__ bias, const float* __restrict__ proj,
    float* __restrict__ scores) {
    __shared__ float sc[32];
    const int tid = threadIdx.x;
    if (tid < 32) sc[tid] = 0.0f;
    __syncthreads();

    const int wave = tid >> 5, lane = tid & 31;
    const int hl = lane >> 4, ln = lane & 15;
    const int r0 = blockIdx.x * 32;
    const int cb = wave * 32;

    const __bf16* b0row = Wt + (size_t)(cb + ln) * 512;
    const __bf16* b1row = Wt + (size_t)(cb + 16 + ln) * 512;

    v8f acc[2][2] = {};
    for (int kk = 0; kk < 16; ++kk) {
        const int e0 = kk * 32;
        v16bf A0 = loadA(rows + (size_t)r0 * 512, 512, 0 + ln, e0, hl);
        v16bf A1 = loadA(rows + (size_t)r0 * 512, 512, 16 + ln, e0, hl);
        v16bf B0 = loadB(b0row, e0, hl);
        v16bf B1 = loadB(b1row, e0, hl);
        acc[0][0] = wmma_bf16(A0, B0, acc[0][0]);
        acc[1][0] = wmma_bf16(A1, B0, acc[1][0]);
        acc[0][1] = wmma_bf16(A0, B1, acc[0][1]);
        acc[1][1] = wmma_bf16(A1, B1, acc[1][1]);
    }
#pragma unroll
    for (int ct = 0; ct < 2; ++ct) {
        int col = cb + ct * 16 + ln;
        float bb = bias[col], pp = proj[col];
#pragma unroll
        for (int rt = 0; rt < 2; ++rt) {
#pragma unroll
            for (int i = 0; i < 8; ++i) {
                float v = tanhf(acc[rt][ct][i] + bb) * pp;
                atomicAdd(&sc[rt * 16 + hl * 8 + i], v);
            }
        }
    }
    __syncthreads();
    if (tid < 32) scores[r0 + tid] = sc[tid];
}

// Softmax over time axis for each (s,b): scores layout [s][T][B]
__global__ void softmax_t(const float* __restrict__ scores,
                          float* __restrict__ a, int T, int nSB) {
    int p = blockIdx.x * blockDim.x + threadIdx.x;
    if (p >= nSB) return;
    int s = p >> 5, b = p & 31;
    const float* base = scores + (size_t)s * T * BN + b;
    float m = -1e30f;
    for (int t = 0; t < T; ++t) m = fmaxf(m, base[t * BN]);
    float sum = 0.0f;
    for (int t = 0; t < T; ++t) sum += __expf(base[t * BN] - m);
    float inv = 1.0f / sum;
    float* ab = a + (size_t)s * T * BN + b;
    for (int t = 0; t < T; ++t) ab[t * BN] = __expf(base[t * BN] - m) * inv;
}

// pooled[s][b][h] = sum_t a[s][t][b] * rows[s][t][b][h]
__global__ void attn_pool(const float* __restrict__ a,
                          const __bf16* __restrict__ rows,
                          __bf16* __restrict__ outp, int nS, int T) {
    int idx = blockIdx.x * blockDim.x + threadIdx.x;
    if (idx >= nS * BN * 512) return;
    int h = idx & 511;
    int b = (idx >> 9) & 31;
    int s = idx >> 14;
    float acc = 0.0f;
    for (int t = 0; t < T; ++t) {
        acc += a[(size_t)(s * T + t) * BN + b] *
               bf2f(rows[((size_t)(s * T + t) * BN + b) * 512 + h]);
    }
    outp[((size_t)s * BN + b) * 512 + h] = f2bf(acc);
}

// logits[b][c] = pooled[b] . fin_W[c] + fin_b[c]
__global__ void final_fc(const __bf16* __restrict__ pooled,
                         const float* __restrict__ finW,
                         const float* __restrict__ finb,
                         float* __restrict__ out) {
    int idx = blockIdx.x * blockDim.x + threadIdx.x;
    if (idx >= BN * NC) return;
    int c = idx % NC, b = idx / NC;
    float acc = finb[c];
    for (int h = 0; h < 512; ++h)
        acc += bf2f(pooled[(size_t)b * 512 + h]) * finW[(size_t)c * 512 + h];
    out[b * NC + c] = acc;
}

// ---------------------------------------------------------------------------
// Launch
// ---------------------------------------------------------------------------
extern "C" void kernel_launch(void* const* d_in, const int* in_sizes, int n_in,
                              void* d_out, int out_size, void* d_ws, size_t ws_size,
                              hipStream_t stream) {
    (void)in_sizes; (void)n_in; (void)out_size; (void)ws_size;

    const int*   x       = (const int*)d_in[0];
    const float* embed   = (const float*)d_in[1];
    const float* wWihF   = (const float*)d_in[2];
    const float* wWhhF   = (const float*)d_in[3];
    const float* wbihF   = (const float*)d_in[4];
    const float* wbhhF   = (const float*)d_in[5];
    const float* wWihB   = (const float*)d_in[6];
    const float* wWhhB   = (const float*)d_in[7];
    const float* wbihB   = (const float*)d_in[8];
    const float* wbhhB   = (const float*)d_in[9];
    const float* word_W  = (const float*)d_in[10];
    const float* word_b  = (const float*)d_in[11];
    const float* word_p  = (const float*)d_in[12];
    const float* sWihF   = (const float*)d_in[13];
    const float* sWhhF   = (const float*)d_in[14];
    const float* sbihF   = (const float*)d_in[15];
    const float* sbhhF   = (const float*)d_in[16];
    const float* sWihB   = (const float*)d_in[17];
    const float* sWhhB   = (const float*)d_in[18];
    const float* sbihB   = (const float*)d_in[19];
    const float* sbhhB   = (const float*)d_in[20];
    const float* sent_W  = (const float*)d_in[21];
    const float* sent_b  = (const float*)d_in[22];
    const float* sent_p  = (const float*)d_in[23];
    const float* fin_W   = (const float*)d_in[24];
    const float* fin_b   = (const float*)d_in[25];
    const float* st_word = (const float*)d_in[26];
    const float* st_sent = (const float*)d_in[27];
    float* out = (float*)d_out;

    // Workspace carve-up (256B aligned)
    char* w = (char*)d_ws;
    size_t off = 0;
    auto alloc = [&](size_t bytes) {
        void* p = w + off;
        off = (off + bytes + 255) & ~(size_t)255;
        return p;
    };
    __bf16* xall   = (__bf16*)alloc((size_t)SN * TN * BN * EN * 2);   // 16 MB
    __bf16* bWihF  = (__bf16*)alloc(768 * 256 * 2);
    __bf16* bWhhF  = (__bf16*)alloc(768 * 256 * 2);
    __bf16* bWihB  = (__bf16*)alloc(768 * 256 * 2);
    __bf16* bWhhB  = (__bf16*)alloc(768 * 256 * 2);
    __bf16* bSihF  = (__bf16*)alloc(768 * 512 * 2);
    __bf16* bShhF  = (__bf16*)alloc(768 * 256 * 2);
    __bf16* bSihB  = (__bf16*)alloc(768 * 512 * 2);
    __bf16* bShhB  = (__bf16*)alloc(768 * 256 * 2);
    __bf16* wordWt = (__bf16*)alloc(512 * 512 * 2);
    __bf16* sentWt = (__bf16*)alloc(512 * 512 * 2);
    __bf16* wout   = (__bf16*)alloc((size_t)SN * TN * BN * 512 * 2);  // 32 MB
    __bf16* svec   = (__bf16*)alloc((size_t)SN * BN * 512 * 2);
    __bf16* sout   = (__bf16*)alloc((size_t)SN * BN * 512 * 2);
    float*  sc_w   = (float*)alloc((size_t)SN * TN * BN * 4);
    float*  a_w    = (float*)alloc((size_t)SN * TN * BN * 4);
    float*  sc_s   = (float*)alloc((size_t)SN * BN * 4);
    float*  a_s    = (float*)alloc((size_t)SN * BN * 4);
    __bf16* pooled = (__bf16*)alloc((size_t)BN * 512 * 2);

    // 1) Weight conversions (bf16; attention W transposed for B-layout)
    auto cvt = [&](const float* s, __bf16* d, int n) {
        cvt_bf16<<<(n + 255) / 256, 256, 0, stream>>>(s, d, n);
    };
    cvt(wWihF, bWihF, 768 * 256);  cvt(wWhhF, bWhhF, 768 * 256);
    cvt(wWihB, bWihB, 768 * 256);  cvt(wWhhB, bWhhB, 768 * 256);
    cvt(sWihF, bSihF, 768 * 512);  cvt(sWhhF, bShhF, 768 * 256);
    cvt(sWihB, bSihB, 768 * 512);  cvt(sWhhB, bShhB, 768 * 256);
    transcvt_bf16<<<1024, 256, 0, stream>>>(word_W, wordWt);
    transcvt_bf16<<<1024, 256, 0, stream>>>(sent_W, sentWt);

    // 2) Embedding gather -> [S,T,B,E] bf16
    gather_embed<<<(SN * TN * BN * (EN / 4)) / 256, 256, 0, stream>>>(x, embed, xall);

    // 3) Word-level bi-GRU: 32 blocks = 16 sentences x 2 directions
    gru_kernel<<<SN * 2, 512, 0, stream>>>(
        xall, bWihF, bWihB, bWhhF, bWhhB,
        wbihF, wbhhF, wbihB, wbhhB, st_word, wout, TN, EN / 32);

    // 4) Word attention
    attn_scores<<<(SN * TN * BN) / 32, 512, 0, stream>>>(wout, wordWt, word_b, word_p, sc_w);
    softmax_t<<<2, 256, 0, stream>>>(sc_w, a_w, TN, SN * BN);
    attn_pool<<<(SN * BN * 512) / 256, 256, 0, stream>>>(a_w, wout, svec, SN, TN);

    // 5) Sentence-level bi-GRU over S=16 "timesteps", input 2H=512
    gru_kernel<<<2, 512, 0, stream>>>(
        svec, bSihF, bSihB, bShhF, bShhB,
        sbihF, sbhhF, sbihB, sbhhB, st_sent, sout, SN, 512 / 32);

    // 6) Sentence attention
    attn_scores<<<(SN * BN) / 32, 512, 0, stream>>>(sout, sentWt, sent_b, sent_p, sc_s);
    softmax_t<<<1, 32, 0, stream>>>(sc_s, a_s, SN, BN);
    attn_pool<<<(BN * 512) / 256, 256, 0, stream>>>(a_s, sout, pooled, 1, SN);

    // 7) Final classifier
    final_fc<<<(BN * NC + 255) / 256, 256, 0, stream>>>(pooled, fin_W, fin_b, out);
}
